// TransNet_46213848105091
// MI455X (gfx1250) — matrix operand order repaired and checked by reading, not compiled, not run on hardware
//
#include <hip/hip_runtime.h>
#include <hip/hip_bf16.h>

// ---------------------------------------------------------------------------
// Problem constants (from the reference)
// ---------------------------------------------------------------------------
#define BG      128
#define MOL_N   40
#define MOL_E   80
#define PROT_N  200
#define PROT_E  1200
#define NM      (BG*MOL_N)    // 5120
#define EM      (BG*MOL_E)    // 10240
#define NP      (BG*PROT_N)   // 25600
#define EP      (BG*PROT_E)   // 153600
#define XD      69
#define XT      33
#define LAT     256
#define OD      128
#define EPS_F   1e-5f

// ---------------------------------------------------------------------------
// Input-pointer indices. Top-level dict in insertion order, then the nested
// "params" pytree flattened with sorted keys at every level (JAX convention).
// ---------------------------------------------------------------------------
enum {
  IN_XMOL = 0, IN_EIM, IN_BATM, IN_XPROT, IN_EIP, IN_BATP, IN_XG,
  P_BN1_B, P_BN1_G, P_BN2_B, P_BN2_G, P_BN3_B, P_BN3_G, P_BN4_B, P_BN4_G,
  P_BN5_B, P_BN5_G,
  P_DC1_KW, P_DC1_KB, P_DC1_QW, P_DC1_QB, P_DC1_SW, P_DC1_SB, P_DC1_VW, P_DC1_VB,
  P_DC2_KW, P_DC2_KB, P_DC2_QW, P_DC2_QB, P_DC2_SW, P_DC2_SB, P_DC2_VW, P_DC2_VB,
  P_DC3_KW, P_DC3_KB, P_DC3_QW, P_DC3_QB, P_DC3_SW, P_DC3_SB, P_DC3_VW, P_DC3_VB,
  P_FC1_W, P_FC1_B, P_FC2_W, P_FC2_B,
  P_FCGD1_W, P_FCGD1_B, P_FCGD2_W, P_FCGD2_B,
  P_FCXT1_W, P_FCXT1_B, P_FCXT2_W, P_FCXT2_B,
  P_GND1_A, P_GND1_Bb, P_GND1_W,
  P_GND2_A, P_GND2_Bb, P_GND2_W,
  P_GND3_A, P_GND3_Bb, P_GND3_W,
  P_GNT1_A, P_GNT1_Bb, P_GNT1_W,
  P_GNT2_A, P_GNT2_Bb, P_GNT2_W,
  P_GNT3_A, P_GNT3_Bb, P_GNT3_W,
  P_GLOB1_W, P_GLOB1_B, P_GLOB2_W, P_GLOB2_B,
  P_OUT_W, P_OUT_B,
  P_TC1_W, P_TC1_B, P_TC2_W, P_TC2_B, P_TC3_W, P_TC3_B
};

// ---------------------------------------------------------------------------
// WMMA GEMM: C[M,N] = act(A[M,K] @ W[K,N] + bias)
// f32 in/out, f16 multiply with f32 accumulate via v_wmma_f32_16x16x32_f16.
// Block = 256 threads (8 wave32). Block tile 128x64, wave tile 32x32
// (2x2 grid of 16x16 accumulators -> 4 WMMA per staged K-step).
// ---------------------------------------------------------------------------
typedef __attribute__((ext_vector_type(16))) _Float16 v16h;
typedef __attribute__((ext_vector_type(8)))  float    v8f;

#define TBM 128
#define TBN 64
#define TBK 32

union H2U  { _Float16 h[2]; unsigned u; };
union FragU { unsigned u[8]; v16h v; };

static __device__ inline unsigned pkh(float a, float b) {
  H2U p; p.h[0] = (_Float16)a; p.h[1] = (_Float16)b; return p.u;
}

__global__ __launch_bounds__(256)
void gemm_wmma_kernel(const float* __restrict__ A, const float* __restrict__ W,
                      const float* __restrict__ bias, float* __restrict__ C,
                      int M, int N, int K, int act)
{
  // A tile: [m][k] f16 pairs (row stride TBK/2=16 uints).
  // B tile: [k][n] f16 pairs (row stride TBN/2=32 uints).
  __shared__ unsigned ldsA[TBM * TBK / 2];   // 8 KB
  __shared__ unsigned ldsB[TBK * TBN / 2];   // 4 KB

  const int tid  = threadIdx.x;
  const int lane = tid & 31;
  const int wv   = tid >> 5;        // 0..7
  const int msub = wv >> 1;         // 0..3 (32-row slab)
  const int nsub = wv & 1;          // 0..1 (32-col slab)
  const int g    = lane >> 4;       // half-wave group
  const int r16  = lane & 15;
  const int m0   = blockIdx.y * TBM;
  const int n0   = blockIdx.x * TBN;

  const bool fullM = (m0 + TBM) <= M;
  const bool fullN = (n0 + TBN) <= N;

  v8f acc00 = {}, acc01 = {}, acc10 = {}, acc11 = {};

  for (int k0 = 0; k0 < K; k0 += TBK) {
    const bool fullK = (k0 + TBK) <= K;

    // ---------------- stage A tile (2048 uints, 8/thread) ------------------
    if (fullM && fullK) {
      for (int s = tid; s < TBM * TBK / 2; s += 256) {
        int m = s >> 4;                 // TBK/2 = 16
        int k = (s & 15) * 2;
        const float* ap = A + (size_t)(m0 + m) * K + k0 + k;
        ldsA[s] = pkh(ap[0], ap[1]);
      }
    } else {
      for (int s = tid; s < TBM * TBK / 2; s += 256) {
        int m  = s >> 4;
        int k  = (s & 15) * 2;
        int gm = m0 + m, gk = k0 + k;
        float f0 = 0.f, f1 = 0.f;
        if (gm < M) {
          if (gk     < K) f0 = A[(size_t)gm * K + gk];
          if (gk + 1 < K) f1 = A[(size_t)gm * K + gk + 1];
        }
        ldsA[s] = pkh(f0, f1);
      }
    }
    // ---------------- stage B tile (1024 uints, 4/thread) ------------------
    if (fullN && fullK) {
      for (int s = tid; s < TBK * TBN / 2; s += 256) {
        int k = s >> 5;                 // TBN/2 = 32
        int n = (s & 31) * 2;
        const float* wp = W + (size_t)(k0 + k) * N + n0 + n;
        ldsB[s] = pkh(wp[0], wp[1]);
      }
    } else {
      for (int s = tid; s < TBK * TBN / 2; s += 256) {
        int k  = s >> 5;
        int n  = (s & 31) * 2;
        int gk = k0 + k, gn = n0 + n;
        float f0 = 0.f, f1 = 0.f;
        if (gk < K) {
          if (gn     < N) f0 = W[(size_t)gk * N + gn];
          if (gn + 1 < N) f1 = W[(size_t)gk * N + gn + 1];
        }
        ldsB[s] = pkh(f0, f1);
      }
    }
    // speculative prefetch of the next K tile (global_prefetch_b8, TH=RT spec)
    if (k0 + TBK < K) {
      __builtin_prefetch(&A[(size_t)(m0 + (tid >> 1)) * K + (k0 + TBK)], 0, 1);
      __builtin_prefetch(&W[(size_t)(k0 + TBK + (tid & 31)) * N + n0], 0, 1);
    }
    __syncthreads();

    // ---------------- fragments per ISA layout ------------------------------
    // A (16x32 f16): lane -> M row, K over VGPR pairs + half-wave group.
    // B (32x16 f16): lane -> N column, K over VGPR pairs + half-wave group.
    FragU fa0, fa1, fb0, fb1;
    const int am0 = msub * 32 + r16;
    const int am1 = am0 + 16;
#pragma unroll
    for (int v = 0; v < 8; ++v) {
      int kb = (v < 4) ? (2 * v + 8 * g) : (16 + 2 * (v - 4) + 8 * g);
      int kh = kb >> 1;
      fa0.u[v] = ldsA[am0 * 16 + kh];
      fa1.u[v] = ldsA[am1 * 16 + kh];
      const _Float16* hB = (const _Float16*)ldsB;
      const int bn0 = nsub * 32 + r16;
      H2U t0, t1;
      t0.h[0] = hB[kb * 64 + bn0];      t0.h[1] = hB[(kb + 1) * 64 + bn0];
      t1.h[0] = hB[kb * 64 + bn0 + 16]; t1.h[1] = hB[(kb + 1) * 64 + bn0 + 16];
      fb0.u[v] = t0.u;
      fb1.u[v] = t1.u;
    }
    acc00 = __builtin_amdgcn_wmma_f32_16x16x32_f16(false, fa0.v, false, fb0.v,
                                                   (short)0, acc00, false, false);
    acc01 = __builtin_amdgcn_wmma_f32_16x16x32_f16(false, fa0.v, false, fb1.v,
                                                   (short)0, acc01, false, false);
    acc10 = __builtin_amdgcn_wmma_f32_16x16x32_f16(false, fa1.v, false, fb0.v,
                                                   (short)0, acc10, false, false);
    acc11 = __builtin_amdgcn_wmma_f32_16x16x32_f16(false, fa1.v, false, fb1.v,
                                                   (short)0, acc11, false, false);
    __syncthreads();
  }

  // ---------------- store C (f32 16x16 layout: lane->N, VGPR+group->M) ------
  const v8f* accs[4] = { &acc00, &acc01, &acc10, &acc11 };
#pragma unroll
  for (int i = 0; i < 2; ++i) {
#pragma unroll
    for (int j = 0; j < 2; ++j) {
      const v8f a = *accs[i * 2 + j];
#pragma unroll
      for (int r = 0; r < 8; ++r) {
        int mm = m0 + msub * 32 + i * 16 + r + 8 * g;
        int nn = n0 + nsub * 32 + j * 16 + r16;
        if (mm < M && nn < N) {
          float v = a[r];
          if (bias) v += bias[nn];
          if (act)  v = fmaxf(v, 0.f);
          C[(size_t)mm * N + nn] = v;
        }
      }
    }
  }
}

// ---------------------------------------------------------------------------
// Elementwise / scatter helpers
// ---------------------------------------------------------------------------
__device__ inline void atomicMaxF(float* addr, float val) {
  int* ai = (int*)addr;
  int old = *ai;
  while (__int_as_float(old) < val) {
    int assumed = old;
    old = atomicCAS(ai, assumed, __float_as_int(val));
    if (old == assumed) break;
  }
}

__global__ void fill_f32(float* p, float v, long long n) {
  long long i = (long long)blockIdx.x * 256 + threadIdx.x;
  if (i < n) p[i] = v;
}

__global__ void bcast_bias(float* out, const float* bias, long long n, int C) {
  long long i = (long long)blockIdx.x * 256 + threadIdx.x;
  if (i < n) out[i] = bias[(int)(i % C)];
}

__global__ void add2_kernel(const float* a, const float* b, float* out, long long n) {
  long long i = (long long)blockIdx.x * 256 + threadIdx.x;
  if (i < n) out[i] = a[i] + b[i];
}

// per-edge, per-head attention logits + running segment max over dst
__global__ void edge_logits_max(const float* __restrict__ q, const float* __restrict__ k,
                                const int* __restrict__ src, const int* __restrict__ dst,
                                int E, int F, float scale,
                                float* __restrict__ logits, float* __restrict__ mmax) {
  int i = blockIdx.x * 256 + threadIdx.x;
  if (i >= E * 2) return;
  int e = i >> 1, h = i & 1;
  int s = src[e], d = dst[e];
  const float* qp = q + (size_t)d * 2 * F + h * F;
  const float* kp = k + (size_t)s * 2 * F + h * F;
  float acc = 0.f;
  for (int f = 0; f < F; ++f) acc += qp[f] * kp[f];
  acc *= scale;
  logits[i] = acc;
  atomicMaxF(&mmax[d * 2 + h], acc);
}

__global__ void edge_exp_den(float* __restrict__ logits, const float* __restrict__ mmax,
                             const int* __restrict__ dst, int E, float* __restrict__ den) {
  int i = blockIdx.x * 256 + threadIdx.x;
  if (i >= E * 2) return;
  int e = i >> 1, h = i & 1;
  int d = dst[e];
  float ex = __expf(logits[i] - mmax[d * 2 + h]);
  logits[i] = ex;
  atomicAdd(&den[d * 2 + h], ex);
}

__global__ void edge_aggregate(const float* __restrict__ ew, const float* __restrict__ den,
                               const float* __restrict__ v,
                               const int* __restrict__ src, const int* __restrict__ dst,
                               int E, int F, float* __restrict__ agg) {
  long long i = (long long)blockIdx.x * 256 + threadIdx.x;
  long long tot = (long long)E * 2 * F;
  if (i >= tot) return;
  int f = (int)(i % F);
  long long t = i / F;
  int h = (int)(t & 1);
  int e = (int)(t >> 1);
  int s = src[e], d = dst[e];
  float att = ew[e * 2 + h] / (den[d * 2 + h] + 1e-16f);
  atomicAdd(&agg[(size_t)d * 2 * F + h * F + f],
            att * v[(size_t)s * 2 * F + h * F + f]);
}

// GraphNorm + optional ReLU. Exploits contiguous per-graph node blocks.
__global__ void graph_norm_relu(const float* __restrict__ x,
                                const float* __restrict__ alpha,
                                const float* __restrict__ w, const float* __restrict__ b,
                                float* __restrict__ out,
                                int nb, int nper, int C, int relu) {
  int i = blockIdx.x * 256 + threadIdx.x;
  if (i >= nb * C) return;
  int gg = i / C, c = i % C;
  const float* xp = x + (size_t)gg * nper * C + c;
  float s = 0.f;
  for (int n = 0; n < nper; ++n) s += xp[(size_t)n * C];
  float am = alpha[c] * (s / nper);
  float vs = 0.f;
  for (int n = 0; n < nper; ++n) { float o = xp[(size_t)n * C] - am; vs += o * o; }
  float inv = rsqrtf(vs / nper + EPS_F);
  float wc = w[c], bc = b[c];
  float* op = out + (size_t)gg * nper * C + c;
  for (int n = 0; n < nper; ++n) {
    float o = (xp[(size_t)n * C] - am) * inv * wc + bc;
    if (relu) o = fmaxf(o, 0.f);
    op[(size_t)n * C] = o;
  }
}

__global__ void deg_count(const int* __restrict__ dst, int E, float* __restrict__ deg) {
  int i = blockIdx.x * 256 + threadIdx.x;
  if (i < E) atomicAdd(&deg[dst[i]], 1.f);
}

__global__ void deg_inv(float* deg, int N) {
  int i = blockIdx.x * 256 + threadIdx.x;
  if (i >= N) return;
  float d = deg[i];
  deg[i] = (d > 0.f) ? rsqrtf(fmaxf(d, 1e-12f)) : 0.f;
}

__global__ void gcn_agg(const float* __restrict__ h, const float* __restrict__ dinv,
                        const int* __restrict__ src, const int* __restrict__ dst,
                        int E, int C, float* __restrict__ out) {
  long long i = (long long)blockIdx.x * 256 + threadIdx.x;
  long long tot = (long long)E * C;
  if (i >= tot) return;
  int c = (int)(i % C);
  int e = (int)(i / C);
  int s = src[e], d = dst[e];
  atomicAdd(&out[(size_t)d * C + c], dinv[s] * dinv[d] * h[(size_t)s * C + c]);
}

__global__ void mean_pool(const float* __restrict__ x, float* __restrict__ out,
                          int nb, int nper, int C) {
  int i = blockIdx.x * 256 + threadIdx.x;
  if (i >= nb * C) return;
  int gg = i / C, c = i % C;
  const float* xp = x + (size_t)gg * nper * C + c;
  float s = 0.f;
  for (int n = 0; n < nper; ++n) s += xp[(size_t)n * C];
  out[(size_t)gg * C + c] = s / nper;
}

// BatchNorm1d (training-mode batch stats, biased var) + ReLU, in-place
__global__ void bn_relu(float* __restrict__ x, const float* __restrict__ gam,
                        const float* __restrict__ bet, int M, int C) {
  int c = blockIdx.x * 256 + threadIdx.x;
  if (c >= C) return;
  float s = 0.f;
  for (int m = 0; m < M; ++m) s += x[(size_t)m * C + c];
  float mean = s / M;
  float vs = 0.f;
  for (int m = 0; m < M; ++m) { float d = x[(size_t)m * C + c] - mean; vs += d * d; }
  float inv = rsqrtf(vs / M + EPS_F);
  float gc = gam[c], bc = bet[c];
  for (int m = 0; m < M; ++m) {
    float o = gc * (x[(size_t)m * C + c] - mean) * inv + bc;
    x[(size_t)m * C + c] = fmaxf(o, 0.f);
  }
}

__global__ void copy_cols(const float* __restrict__ src, float* __restrict__ dst,
                          int M, int srcC, int dstC, int col0) {
  long long i = (long long)blockIdx.x * 256 + threadIdx.x;
  long long tot = (long long)M * srcC;
  if (i >= tot) return;
  int m = (int)(i / srcC), c = (int)(i % srcC);
  dst[(size_t)m * dstC + col0 + c] = src[i];
}

__global__ void final_out(const float* __restrict__ x, const float* __restrict__ W,
                          const float* __restrict__ b, float* __restrict__ out,
                          int M, int K) {
  int m = blockIdx.x * 256 + threadIdx.x;
  if (m >= M) return;
  float s = b[0];
  for (int k = 0; k < K; ++k) s += x[(size_t)m * K + k] * W[k];
  out[m] = s;
}

// ---------------------------------------------------------------------------
// Host-side orchestration
// ---------------------------------------------------------------------------
static inline int cdiv(long long a, long long b) { return (int)((a + b - 1) / b); }

static void gemm(hipStream_t s, const float* A, const float* W, const float* bias,
                 float* C, int M, int N, int K, int act) {
  dim3 grid(cdiv(N, TBN), cdiv(M, TBM)), blk(256);
  gemm_wmma_kernel<<<grid, blk, 0, s>>>(A, W, bias, C, M, N, K, act);
}
static void fill(hipStream_t s, float* p, float v, long long n) {
  fill_f32<<<cdiv(n, 256), 256, 0, s>>>(p, v, n);
}

static void transformer_layer(hipStream_t s, const float* xin, int Cin, int F,
                              const float* Wq, const float* bq,
                              const float* Wk, const float* bk,
                              const float* Wv, const float* bv,
                              const float* Wsk, const float* bsk,
                              const int* src, const int* dst, int E, int N, int nper,
                              const float* ga, const float* gw, const float* gb,
                              float* Dq, float* Dk, float* Dv, float* Dsk,
                              float* Dagg, float* Dlog, float* Dm, float* Dden,
                              float* xout) {
  const int Cout = 2 * F;
  gemm(s, xin, Wq,  bq,  Dq,  N, Cout, Cin, 0);
  gemm(s, xin, Wk,  bk,  Dk,  N, Cout, Cin, 0);
  gemm(s, xin, Wv,  bv,  Dv,  N, Cout, Cin, 0);
  gemm(s, xin, Wsk, bsk, Dsk, N, Cout, Cin, 0);
  fill(s, Dm,  -__builtin_inff(), (long long)N * 2);
  fill(s, Dden, 0.f,              (long long)N * 2);
  float scale = rsqrtf((float)F);
  edge_logits_max<<<cdiv((long long)E * 2, 256), 256, 0, s>>>(Dq, Dk, src, dst, E, F,
                                                              scale, Dlog, Dm);
  edge_exp_den<<<cdiv((long long)E * 2, 256), 256, 0, s>>>(Dlog, Dm, dst, E, Dden);
  fill(s, Dagg, 0.f, (long long)N * Cout);
  edge_aggregate<<<cdiv((long long)E * 2 * F, 256), 256, 0, s>>>(Dlog, Dden, Dv, src,
                                                                 dst, E, F, Dagg);
  add2_kernel<<<cdiv((long long)N * Cout, 256), 256, 0, s>>>(Dagg, Dsk, xout,
                                                             (long long)N * Cout);
  graph_norm_relu<<<cdiv(BG * Cout, 256), 256, 0, s>>>(xout, ga, gw, gb, xout,
                                                       BG, nper, Cout, 1);
}

static void gcn_layer(hipStream_t s, const float* xin, int Cin, int Cout,
                      const float* W, const float* b,
                      const int* src, const int* dst, int E, int N, int nper,
                      const float* ga, const float* gw, const float* gb,
                      const float* dinv, float* Ph, float* Pagg, float* xout) {
  gemm(s, xin, W, nullptr, Ph, N, Cout, Cin, 0);
  bcast_bias<<<cdiv((long long)N * Cout, 256), 256, 0, s>>>(Pagg, b,
                                                            (long long)N * Cout, Cout);
  gcn_agg<<<cdiv((long long)E * Cout, 256), 256, 0, s>>>(Ph, dinv, src, dst, E,
                                                         Cout, Pagg);
  graph_norm_relu<<<cdiv(BG * Cout, 256), 256, 0, s>>>(Pagg, ga, gw, gb, xout,
                                                       BG, nper, Cout, 1);
}

extern "C" void kernel_launch(void* const* d_in, const int* in_sizes, int n_in,
                              void* d_out, int out_size, void* d_ws, size_t ws_size,
                              hipStream_t stream) {
  (void)in_sizes; (void)n_in; (void)out_size; (void)ws_size;
  const float* x_mol  = (const float*)d_in[IN_XMOL];
  const int*   eim    = (const int*)d_in[IN_EIM];
  const int*   srcm   = eim;
  const int*   dstm   = eim + EM;
  const float* x_prot = (const float*)d_in[IN_XPROT];
  const int*   eip    = (const int*)d_in[IN_EIP];
  const int*   srcp   = eip;
  const int*   dstp   = eip + EP;
  const float* x_glob = (const float*)d_in[IN_XG];
  auto P = [&](int i) -> const float* { return (const float*)d_in[i]; };

  // ---- workspace bump allocator (256B aligned) ----------------------------
  char*  base = (char*)d_ws;
  size_t off  = 0;
  auto alloc = [&](size_t elems) -> float* {
    float* p = (float*)(base + off);
    off += ((elems * sizeof(float) + 255) / 256) * 256;
    return p;
  };

  // drug branch buffers (max channels 552)
  float* Dq   = alloc((size_t)NM * 552);
  float* Dk   = alloc((size_t)NM * 552);
  float* Dv   = alloc((size_t)NM * 552);
  float* Dsk  = alloc((size_t)NM * 552);
  float* Dagg = alloc((size_t)NM * 552);
  float* Dx   = alloc((size_t)NM * 552);
  float* Dlog = alloc((size_t)EM * 2);
  float* Dm   = alloc((size_t)NM * 2);
  float* Dden = alloc((size_t)NM * 2);
  // protein branch buffers (max channels 1024)
  float* Ph    = alloc((size_t)NP * 1024);
  float* Pagg  = alloc((size_t)NP * 1024);
  float* Px    = alloc((size_t)NP * 1024);
  float* Pdinv = alloc((size_t)NP);
  // heads
  float* poolD = alloc((size_t)BG * 552);
  float* gd1   = alloc((size_t)BG * 1024);
  float* xdrug = alloc((size_t)BG * OD);
  float* poolP = alloc((size_t)BG * 1024);
  float* xg1   = alloc((size_t)BG * 1024);
  float* xg2   = alloc((size_t)BG * 1024);
  float* catP  = alloc((size_t)BG * 2048);
  float* xt1   = alloc((size_t)BG * 1024);
  float* xtf   = alloc((size_t)BG * OD);
  float* cat2  = alloc((size_t)BG * 2 * OD);
  float* f1    = alloc((size_t)BG * 1024);
  float* f2    = alloc((size_t)BG * 512);

  // ======================= drug branch (TransformerConv x3) ================
  transformer_layer(stream, x_mol, XD, XD,
                    P(P_DC1_QW), P(P_DC1_QB), P(P_DC1_KW), P(P_DC1_KB),
                    P(P_DC1_VW), P(P_DC1_VB), P(P_DC1_SW), P(P_DC1_SB),
                    srcm, dstm, EM, NM, MOL_N,
                    P(P_GND1_A), P(P_GND1_W), P(P_GND1_Bb),
                    Dq, Dk, Dv, Dsk, Dagg, Dlog, Dm, Dden, Dx);
  transformer_layer(stream, Dx, 2 * XD, 2 * XD,
                    P(P_DC2_QW), P(P_DC2_QB), P(P_DC2_KW), P(P_DC2_KB),
                    P(P_DC2_VW), P(P_DC2_VB), P(P_DC2_SW), P(P_DC2_SB),
                    srcm, dstm, EM, NM, MOL_N,
                    P(P_GND2_A), P(P_GND2_W), P(P_GND2_Bb),
                    Dq, Dk, Dv, Dsk, Dagg, Dlog, Dm, Dden, Dx);
  transformer_layer(stream, Dx, 4 * XD, 4 * XD,
                    P(P_DC3_QW), P(P_DC3_QB), P(P_DC3_KW), P(P_DC3_KB),
                    P(P_DC3_VW), P(P_DC3_VB), P(P_DC3_SW), P(P_DC3_SB),
                    srcm, dstm, EM, NM, MOL_N,
                    P(P_GND3_A), P(P_GND3_W), P(P_GND3_Bb),
                    Dq, Dk, Dv, Dsk, Dagg, Dlog, Dm, Dden, Dx);
  mean_pool<<<cdiv(BG * 8 * XD, 256), 256, 0, stream>>>(Dx, poolD, BG, MOL_N, 8 * XD);
  gemm(stream, poolD, P(P_FCGD1_W), P(P_FCGD1_B), gd1, BG, 1024, 8 * XD, 0);
  bn_relu<<<cdiv(1024, 256), 256, 0, stream>>>(gd1, P(P_BN1_G), P(P_BN1_B), BG, 1024);
  gemm(stream, gd1, P(P_FCGD2_W), P(P_FCGD2_B), xdrug, BG, OD, 1024, 1);

  // ======================= protein branch (GCN x3) =========================
  fill(stream, Pdinv, 0.f, NP);
  deg_count<<<cdiv(EP, 256), 256, 0, stream>>>(dstp, EP, Pdinv);
  deg_inv<<<cdiv(NP, 256), 256, 0, stream>>>(Pdinv, NP);
  gcn_layer(stream, x_prot, XT, LAT, P(P_TC1_W), P(P_TC1_B),
            srcp, dstp, EP, NP, PROT_N,
            P(P_GNT1_A), P(P_GNT1_W), P(P_GNT1_Bb), Pdinv, Ph, Pagg, Px);
  gcn_layer(stream, Px, LAT, 2 * LAT, P(P_TC2_W), P(P_TC2_B),
            srcp, dstp, EP, NP, PROT_N,
            P(P_GNT2_A), P(P_GNT2_W), P(P_GNT2_Bb), Pdinv, Ph, Pagg, Px);
  gcn_layer(stream, Px, 2 * LAT, 4 * LAT, P(P_TC3_W), P(P_TC3_B),
            srcp, dstp, EP, NP, PROT_N,
            P(P_GNT3_A), P(P_GNT3_W), P(P_GNT3_Bb), Pdinv, Ph, Pagg, Px);
  mean_pool<<<cdiv(BG * 4 * LAT, 256), 256, 0, stream>>>(Px, poolP, BG, PROT_N, 4 * LAT);

  // global features
  gemm(stream, x_glob, P(P_GLOB1_W), P(P_GLOB1_B), xg1, BG, 1024, XT, 0);
  bn_relu<<<cdiv(1024, 256), 256, 0, stream>>>(xg1, P(P_BN5_G), P(P_BN5_B), BG, 1024);
  gemm(stream, xg1, P(P_GLOB2_W), P(P_GLOB2_B), xg2, BG, 4 * LAT, 1024, 1);

  // concat [poolP | xg2] -> fc_xt1 -> bn2+relu -> fc_xt2+relu
  copy_cols<<<cdiv((long long)BG * 1024, 256), 256, 0, stream>>>(poolP, catP, BG, 1024, 2048, 0);
  copy_cols<<<cdiv((long long)BG * 1024, 256), 256, 0, stream>>>(xg2,   catP, BG, 1024, 2048, 1024);
  gemm(stream, catP, P(P_FCXT1_W), P(P_FCXT1_B), xt1, BG, 1024, 2048, 0);
  bn_relu<<<cdiv(1024, 256), 256, 0, stream>>>(xt1, P(P_BN2_G), P(P_BN2_B), BG, 1024);
  gemm(stream, xt1, P(P_FCXT2_W), P(P_FCXT2_B), xtf, BG, OD, 1024, 1);

  // ======================= fusion head =====================================
  copy_cols<<<cdiv((long long)BG * OD, 256), 256, 0, stream>>>(xdrug, cat2, BG, OD, 2 * OD, 0);
  copy_cols<<<cdiv((long long)BG * OD, 256), 256, 0, stream>>>(xtf,   cat2, BG, OD, 2 * OD, OD);
  gemm(stream, cat2, P(P_FC1_W), P(P_FC1_B), f1, BG, 1024, 2 * OD, 0);
  bn_relu<<<cdiv(1024, 256), 256, 0, stream>>>(f1, P(P_BN3_G), P(P_BN3_B), BG, 1024);
  gemm(stream, f1, P(P_FC2_W), P(P_FC2_B), f2, BG, 512, 1024, 0);
  bn_relu<<<cdiv(512, 256), 256, 0, stream>>>(f2, P(P_BN4_G), P(P_BN4_B), BG, 512);
  final_out<<<cdiv(BG, 256), 256, 0, stream>>>(f2, P(P_OUT_W), P(P_OUT_B),
                                               (float*)d_out, BG, 512);
}
